// IEKTNet_49426483642716
// MI455X (gfx1250) — matrix operand order repaired and checked by reading, not compiled, not run on hardware
//
#include <hip/hip_runtime.h>
#include <hip/hip_bf16.h>

// ---------------------------------------------------------------------------
// IEKT forward on MI455X (gfx1250, wave32, WMMA).
//  - 32 workgroups x 16 batch rows; each WG runs the full 128-step scan.
//  - All GEMMs via v_wmma_f32_16x16x32_bf16 (M=16 tiles, waves own N-tiles).
//  - Weights repacked once per launch into the wave32 WMMA B-operand layout.
//  - softmax dropped (argmax-equivalent); gt/op in {0,1} -> mux copies.
// ---------------------------------------------------------------------------

typedef __bf16 bf16_t;
typedef __attribute__((ext_vector_type(16))) __bf16 v16bf;
typedef __attribute__((ext_vector_type(8)))  __bf16 v8bf;
typedef __attribute__((ext_vector_type(8)))  float  v8f;

#define E_   64
#define B_   512
#define S_   128
#define NWAVE 8

// packed-B element offsets (bf16 elements) inside d_ws
#define OFF_SELW1  0u          // 6*12*512  = 36864
#define OFF_SELWO  36864u      // 6*1*512   = 3072
#define OFF_PREDW1 39936u      // 10*20*512 = 102400
#define OFF_CHKW1  142336u     // 24*48*512 = 589824
#define OFF_CHKWO  732160u     // 24*1*512  = 12288
#define OFF_GRUWI  744448u     // 8*12*512  = 49152
#define OFF_GRUWH  793600u     // 2*12*512  = 12288
#define WS_ELEMS   805888u

union V16U { v16bf v; v8bf h[2]; };

// A operand (16x32 bf16) from LDS row-major [16 x lda]:
// lane 0-15: row=lane, K = 0..7 and 16..23 ; lane 16-31: row=lane-16, K = 8..15 and 24..31
static __device__ __forceinline__ v16bf load_a_tile(const bf16_t* A, int lda, int kt, int lane) {
  int row = lane & 15;
  int kb  = (lane >> 4) * 8;
  const bf16_t* p = A + row * lda + kt * 32 + kb;
  V16U u;
  u.h[0] = *(const v8bf*)(p);
  u.h[1] = *(const v8bf*)(p + 16);
  return u.v;
}

// B operand (32x16 bf16) pre-packed: tile (kt,nt), 32 lanes x 16 contiguous bf16.
static __device__ __forceinline__ v16bf load_b_tile(const bf16_t* Bp, int kt, int nt, int Nt, int lane) {
  return *(const v16bf*)(Bp + ((size_t)((kt * Nt + nt) * 32 + lane)) * 16);
}

static __device__ __forceinline__ v8f gemm_acc(const bf16_t* A, int lda, const bf16_t* Bp,
                                               int Kt, int nt, int Nt, int lane) {
  v8f acc = {};
  for (int kt = 0; kt < Kt; ++kt) {
    v16bf a = load_a_tile(A, lda, kt, lane);
    v16bf b = load_b_tile(Bp, kt, nt, Nt, lane);
    acc = __builtin_amdgcn_wmma_f32_16x16x32_bf16(false, a, false, b, (short)0, acc, false, false);
  }
  return acc;
}

// C/D layout: lane -> col = lane&15 ; VGPR i -> row = i + (lane>>4)*8
static __device__ __forceinline__ void gemm_relu(const bf16_t* A, int lda, const bf16_t* Bp,
                                                 int Kt, int Nt, const float* bias,
                                                 bf16_t* Out, int ldo, int lane, int wave) {
  for (int nt = wave; nt < Nt; nt += NWAVE) {
    v8f acc = gemm_acc(A, lda, Bp, Kt, nt, Nt, lane);
    int col = lane & 15, rbase = (lane >> 4) * 8, n = nt * 16 + col;
    float bb = bias[n];
    #pragma unroll
    for (int i = 0; i < 8; ++i) {
      float x = acc[i] + bb;
      Out[(rbase + i) * ldo + n] = (bf16_t)fmaxf(x, 0.0f);
    }
  }
}

// Single N-tile head producing f32 logits [16 x 16] (cols >= nbias padded to -inf)
static __device__ __forceinline__ void gemm_logits(const bf16_t* A, int lda, const bf16_t* Bp,
                                                   int Kt, const float* bias, int nbias,
                                                   float* OutF, int lane, int wave) {
  if (wave == 0) {
    v8f acc = gemm_acc(A, lda, Bp, Kt, 0, 1, lane);
    int col = lane & 15, rbase = (lane >> 4) * 8;
    float bb = (col < nbias) ? bias[col] : -3.0e38f;
    #pragma unroll
    for (int i = 0; i < 8; ++i)
      OutF[(rbase + i) * 16 + col] = acc[i] + bb;
  }
}

static __device__ __forceinline__ float sigmoidf_(float x) {
  return 1.0f / (1.0f + __expf(-x));
}

// --------------------- weight repacking (once per launch) -------------------
// packed[((kt*Nt+nt)*32 + lane)*16 + j] = W[K][N] with
//   K = kt*32 + (lane>>4)*8 + (j&7) + (j>>3)*16 , N = nt*16 + (lane&15) ; OOB -> 0
__global__ void pack_b_kernel(const float* __restrict__ W, bf16_t* __restrict__ out,
                              int Kdim, int Ncols, int Nt, int total) {
  int idx = blockIdx.x * blockDim.x + threadIdx.x;
  if (idx >= total) return;
  int j    = idx & 15;
  int lane = (idx >> 4) & 31;
  int tIdx = idx >> 9;
  int kt = tIdx / Nt, nt = tIdx - kt * Nt;
  int K = kt * 32 + (lane >> 4) * 8 + (j & 7) + (j >> 3) * 16;
  int N = nt * 16 + (lane & 15);
  float v = (K < Kdim && N < Ncols) ? W[(size_t)K * Ncols + N] : 0.0f;
  out[idx] = (bf16_t)v;
}

// fuse three [Kdim,64] gate matrices into one packed [Kdim,192] operand
__global__ void pack3_kernel(const float* __restrict__ W0, const float* __restrict__ W1,
                             const float* __restrict__ W2, bf16_t* __restrict__ out,
                             int Kdim, int total) {
  const int Nt = 12;
  int idx = blockIdx.x * blockDim.x + threadIdx.x;
  if (idx >= total) return;
  int j    = idx & 15;
  int lane = (idx >> 4) & 31;
  int tIdx = idx >> 9;
  int kt = tIdx / Nt, nt = tIdx - kt * Nt;
  int K = kt * 32 + (lane >> 4) * 8 + (j & 7) + (j >> 3) * 16;
  int N = nt * 16 + (lane & 15);
  float v = 0.0f;
  if (K < Kdim) {
    const float* W = (N < 64) ? W0 : (N < 128) ? W1 : W2;
    v = W[(size_t)K * 64 + (N & 63)];
  }
  out[idx] = (bf16_t)v;
}

// ------------------------------- main kernel --------------------------------
struct IektParams {
  const int* cq; const int* cc; const int* cr;
  const float* que; const float* con; const float* cog; const float* acqm;
  const float* sel_b1; const float* sel_bo;
  const float* pred_b1; const float* pred_Wo; const float* pred_bo;
  const float* chk_b1; const float* chk_bo;
  const float* b_ir; const float* b_iz; const float* b_in;
  const float* b_hr; const float* b_hz; const float* b_hn;
  const bf16_t* selW1p; const bf16_t* selWop;
  const bf16_t* predW1p; const bf16_t* chkW1p; const bf16_t* chkWop;
  const bf16_t* gruWip; const bf16_t* gruWhp;
  float* out;
};

__global__ __launch_bounds__(256) void iekt_main(IektParams p) {
  __shared__ alignas(32) bf16_t sh_x[16 * 768];   // GEMM inputs (mux-built)
  __shared__ alignas(32) bf16_t sh_a[16 * 768];   // hidden activations
  __shared__ alignas(32) bf16_t sh_v[16 * 128];   // per-step question repr v
  __shared__ alignas(32) bf16_t sh_hb[16 * 64];   // h (bf16, WMMA input)
  __shared__ alignas(32) float  sh_h[16 * 64];    // h (f32, recurrent state)
  __shared__ alignas(32) bf16_t sh_rg[16 * 64];   // GRU r gate
  __shared__ alignas(32) bf16_t sh_zg[16 * 64];   // GRU z gate
  __shared__ float sh_lg[16 * 16];                // logits / reduction scratch
  __shared__ int sh_gt[16], sh_op[16], sh_i1[16], sh_i2[16];

  const int tid  = threadIdx.x;
  const int lane = tid & 31, wave = tid >> 5;
  const int col  = lane & 15, rbase = (lane >> 4) * 8;
  const int b0   = blockIdx.x * 16;

  for (int i = tid; i < 16 * 64; i += 256) { sh_h[i] = 0.0f; sh_hb[i] = (bf16_t)0.0f; }
  __syncthreads();

  for (int t = 0; t < S_; ++t) {
    // ---- P0: v = [mean of 4 concept rows | que row]; gt = cr ----
    for (int i = tid; i < 16 * 64; i += 256) {
      int r = i >> 6, e = i & 63, b = b0 + r;
      const int* cp = p.cc + (size_t)(b * S_ + t) * 4;
      float s = p.con[(size_t)cp[0] * 64 + e] + p.con[(size_t)cp[1] * 64 + e]
              + p.con[(size_t)cp[2] * 64 + e] + p.con[(size_t)cp[3] * 64 + e];
      sh_v[r * 128 + e] = (bf16_t)(s * 0.25f);
      int q = p.cq[b * S_ + t];
      sh_v[r * 128 + 64 + e] = (bf16_t)p.que[(size_t)q * 64 + e];
    }
    if (tid < 16) sh_gt[tid] = p.cr[(b0 + tid) * S_ + t];
    __syncthreads();

    // ---- P1: sel MLP on [v | h] (192) -> argmax -> cog row ----
    for (int i = tid; i < 16 * 192; i += 256) {
      int r = i / 192, c = i - r * 192;
      sh_x[r * 768 + c] = (c < 128) ? sh_v[r * 128 + c] : sh_hb[r * 64 + (c - 128)];
    }
    __syncthreads();
    gemm_relu(sh_x, 768, p.selW1p, 6, 12, p.sel_b1, sh_a, 768, lane, wave);
    __syncthreads();
    gemm_logits(sh_a, 768, p.selWop, 6, p.sel_bo, 10, sh_lg, lane, wave);
    __syncthreads();
    if (tid < 16) {
      float best = sh_lg[tid * 16]; int bi = 0;
      for (int j = 1; j < 10; ++j) { float v = sh_lg[tid * 16 + j]; if (v > best) { best = v; bi = j; } }
      sh_i1[tid] = bi;
    }
    __syncthreads();

    // ---- P2: pred MLP on [h | v | cog_emb] (320) -> prob, op ----
    for (int i = tid; i < 16 * 320; i += 256) {
      int r = i / 320, c = i - r * 320;
      bf16_t v;
      if (c < 64)        v = sh_hb[r * 64 + c];
      else if (c < 192)  v = sh_v[r * 128 + (c - 64)];
      else               v = (bf16_t)p.cog[(size_t)sh_i1[r] * 128 + (c - 192)];
      sh_x[r * 768 + c] = v;
    }
    __syncthreads();
    gemm_relu(sh_x, 768, p.predW1p, 10, 20, p.pred_b1, sh_a, 768, lane, wave);
    __syncthreads();
    {
      int r = tid >> 4, l = tid & 15;
      float part = 0.0f;
      for (int n = l; n < 320; n += 16) part += (float)sh_a[r * 768 + n] * p.pred_Wo[n];
      sh_lg[r * 16 + l] = part;
    }
    __syncthreads();
    if (tid < 16) {
      float s = p.pred_bo[0];
      for (int j = 0; j < 16; ++j) s += sh_lg[tid * 16 + j];
      float pr = sigmoidf_(s);
      sh_op[tid] = (pr > 0.5f) ? 1 : 0;
      p.out[(size_t)(b0 + tid) * S_ + t] = pr;
    }
    __syncthreads();

    // ---- P3: chk MLP on out_x (768, pure mux of h_v by gt/op) -> acq row ----
    for (int i = tid; i < 16 * 768; i += 256) {
      int r = i / 768, c = i - r * 768;
      int seg = c / 192, c2 = c - seg * 192;
      bf16_t hv = (c2 < 64) ? sh_hb[r * 64 + c2] : sh_v[r * 128 + (c2 - 64)];
      int g = sh_gt[r], o = sh_op[r];
      int sel = (seg == 0) ? g : (seg == 1) ? (1 - g) : (seg == 2) ? o : (1 - o);
      sh_x[r * 768 + c] = sel ? hv : (bf16_t)0.0f;
    }
    __syncthreads();
    gemm_relu(sh_x, 768, p.chkW1p, 24, 48, p.chk_b1, sh_a, 768, lane, wave);
    __syncthreads();
    gemm_logits(sh_a, 768, p.chkWop, 24, p.chk_bo, 10, sh_lg, lane, wave);
    __syncthreads();
    if (tid < 16) {
      float best = sh_lg[tid * 16]; int bi = 0;
      for (int j = 1; j < 10; ++j) { float v = sh_lg[tid * 16 + j]; if (v > best) { best = v; bi = j; } }
      sh_i2[tid] = bi;
    }
    __syncthreads();

    // ---- P4: GRU. inp = mux(v, acq_emb by gt) (256) ----
    for (int i = tid; i < 16 * 256; i += 256) {
      int r = i >> 8, c = i & 255, c2 = c & 127;
      int g = sh_gt[r];
      float vv = (float)sh_v[r * 128 + c2];
      float ee = p.acqm[(size_t)sh_i2[r] * 128 + c2];
      float val = (c < 128) ? (g ? vv : ee) : (g ? ee : vv);
      sh_x[r * 768 + c] = (bf16_t)val;
    }
    __syncthreads();

    // pass 1: tiles 0..7 -> r (cols 0..63) and z (cols 64..127)
    {
      int nt = wave;
      v8f accA = gemm_acc(sh_x, 768, p.gruWip, 8, nt, 12, lane);
      v8f accB = gemm_acc(sh_hb, 64, p.gruWhp, 2, nt, 12, lane);
      int n = nt * 16 + col;
      if (n < 64) {
        float bb = p.b_ir[n] + p.b_hr[n];
        #pragma unroll
        for (int i = 0; i < 8; ++i)
          sh_rg[(rbase + i) * 64 + n] = (bf16_t)sigmoidf_(accA[i] + accB[i] + bb);
      } else {
        int c2 = n - 64;
        float bb = p.b_iz[c2] + p.b_hz[c2];
        #pragma unroll
        for (int i = 0; i < 8; ++i)
          sh_zg[(rbase + i) * 64 + c2] = (bf16_t)sigmoidf_(accA[i] + accB[i] + bb);
      }
    }
    // pass 2 (waves 0-3): n-gate tiles 8..11, keep accumulators in registers
    v8f accAn = {}; v8f accBn = {};
    if (wave < 4) {
      int nt = 8 + wave;
      accAn = gemm_acc(sh_x, 768, p.gruWip, 8, nt, 12, lane);
      accBn = gemm_acc(sh_hb, 64, p.gruWhp, 2, nt, 12, lane);
    }
    __syncthreads();
    if (wave < 4) {
      int c2 = wave * 16 + col;
      float bi_ = p.b_in[c2], bh_ = p.b_hn[c2];
      #pragma unroll
      for (int i = 0; i < 8; ++i) {
        int row = rbase + i;
        float rr = (float)sh_rg[row * 64 + c2];
        float nn = tanhf(accAn[i] + bi_ + rr * (accBn[i] + bh_));
        float zz = (float)sh_zg[row * 64 + c2];
        float hn = (1.0f - zz) * nn + zz * sh_h[row * 64 + c2];
        sh_h[row * 64 + c2] = hn;
        sh_hb[row * 64 + c2] = (bf16_t)hn;
      }
    }
    __syncthreads();
  }
}

// ------------------------------- host launch --------------------------------
extern "C" void kernel_launch(void* const* d_in, const int* in_sizes, int n_in,
                              void* d_out, int out_size, void* d_ws, size_t ws_size,
                              hipStream_t stream) {
  (void)in_sizes; (void)n_in; (void)out_size; (void)ws_size;
  const int*   cq  = (const int*)d_in[0];
  const int*   cc  = (const int*)d_in[1];
  const int*   cr  = (const int*)d_in[2];
  const float* que = (const float*)d_in[3];
  const float* con = (const float*)d_in[4];
  const float* cog = (const float*)d_in[5];
  const float* acqm= (const float*)d_in[6];
  const float* sel_W1 = (const float*)d_in[7];   const float* sel_b1 = (const float*)d_in[8];
  const float* sel_Wo = (const float*)d_in[9];   const float* sel_bo = (const float*)d_in[10];
  const float* pred_W1= (const float*)d_in[11];  const float* pred_b1= (const float*)d_in[12];
  const float* pred_Wo= (const float*)d_in[13];  const float* pred_bo= (const float*)d_in[14];
  const float* chk_W1 = (const float*)d_in[15];  const float* chk_b1 = (const float*)d_in[16];
  const float* chk_Wo = (const float*)d_in[17];  const float* chk_bo = (const float*)d_in[18];
  const float* W_ir = (const float*)d_in[19];    const float* b_ir = (const float*)d_in[20];
  const float* W_iz = (const float*)d_in[21];    const float* b_iz = (const float*)d_in[22];
  const float* W_in = (const float*)d_in[23];    const float* b_in = (const float*)d_in[24];
  const float* W_hr = (const float*)d_in[25];    const float* b_hr = (const float*)d_in[26];
  const float* W_hz = (const float*)d_in[27];    const float* b_hz = (const float*)d_in[28];
  const float* W_hn = (const float*)d_in[29];    const float* b_hn = (const float*)d_in[30];

  bf16_t* ws = (bf16_t*)d_ws;
  bf16_t* selW1p  = ws + OFF_SELW1;
  bf16_t* selWop  = ws + OFF_SELWO;
  bf16_t* predW1p = ws + OFF_PREDW1;
  bf16_t* chkW1p  = ws + OFF_CHKW1;
  bf16_t* chkWop  = ws + OFF_CHKWO;
  bf16_t* gruWip  = ws + OFF_GRUWI;
  bf16_t* gruWhp  = ws + OFF_GRUWH;

  auto blocks = [](int total) { return (total + 255) / 256; };

  pack_b_kernel<<<blocks(6 * 12 * 512), 256, 0, stream>>>(sel_W1, selW1p, 192, 192, 12, 6 * 12 * 512);
  pack_b_kernel<<<blocks(6 * 1 * 512), 256, 0, stream>>>(sel_Wo, selWop, 192, 10, 1, 6 * 1 * 512);
  pack_b_kernel<<<blocks(10 * 20 * 512), 256, 0, stream>>>(pred_W1, predW1p, 320, 320, 20, 10 * 20 * 512);
  pack_b_kernel<<<blocks(24 * 48 * 512), 256, 0, stream>>>(chk_W1, chkW1p, 768, 768, 48, 24 * 48 * 512);
  pack_b_kernel<<<blocks(24 * 1 * 512), 256, 0, stream>>>(chk_Wo, chkWop, 768, 10, 1, 24 * 1 * 512);
  pack3_kernel<<<blocks(8 * 12 * 512), 256, 0, stream>>>(W_ir, W_iz, W_in, gruWip, 256, 8 * 12 * 512);
  pack3_kernel<<<blocks(2 * 12 * 512), 256, 0, stream>>>(W_hr, W_hz, W_hn, gruWhp, 64, 2 * 12 * 512);

  IektParams p;
  p.cq = cq; p.cc = cc; p.cr = cr;
  p.que = que; p.con = con; p.cog = cog; p.acqm = acqm;
  p.sel_b1 = sel_b1; p.sel_bo = sel_bo;
  p.pred_b1 = pred_b1; p.pred_Wo = pred_Wo; p.pred_bo = pred_bo;
  p.chk_b1 = chk_b1; p.chk_bo = chk_bo;
  p.b_ir = b_ir; p.b_iz = b_iz; p.b_in = b_in;
  p.b_hr = b_hr; p.b_hz = b_hz; p.b_hn = b_hn;
  p.selW1p = selW1p; p.selWop = selWop;
  p.predW1p = predW1p; p.chkW1p = chkW1p; p.chkWop = chkWop;
  p.gruWip = gruWip; p.gruWhp = gruWhp;
  p.out = (float*)d_out;

  iekt_main<<<dim3(B_ / 16), dim3(256), 0, stream>>>(p);
}
